// ContentTemporalAttention_4389456577360
// MI455X (gfx1250) — matrix operand order repaired and checked by reading, not compiled, not run on hardware
//
#include <hip/hip_runtime.h>
#include <hip/hip_bf16.h>

// ---------------------------------------------------------------------------
// ContentTemporalAttention for MI455X (gfx1250, wave32, WMMA)
//   B=4, S=1024, D=512, H=8, Hd=64, TB_HID=32
// Pipeline:
//   1) cvt inputs -> bf16
//   2) transpose+cvt weights -> bf16 (W^T so B-operand loads are contiguous)
//   3) WMMA GEMM: q,k -> [B,H,S,Hd] bf16 ; v -> [B,H,Hd,S] bf16 (transposed)
//   4) temporal-bias MLP -> bf16 [B,H,S,S] (stays in 192MB L2)
//   5) flash-attention per (b,h,qtile): v_wmma_f32_16x16x32_bf16 for QK^T and PV,
//      online softmax in C/D layout, P relayout via per-wave LDS
//   6) WMMA GEMM: ctx @ Wo + bo -> f32 d_out
// ---------------------------------------------------------------------------

typedef __attribute__((ext_vector_type(16))) __bf16 v16bf;
typedef __attribute__((ext_vector_type(8)))  float  v8f;

#define D_MODEL 512
#define SEQ     1024
#define NBATCH  4
#define NH      8
#define HD      64

union BFPack { v16bf v; uint4 q[2]; };

// A-operand (16x32 bf16): per lane two 16B chunks: K = g*8..g*8+7 and +16..+23
__device__ __forceinline__ v16bf ld_a_op(const __bf16* p0, const __bf16* p1) {
  BFPack r;
  r.q[0] = *(const uint4*)p0;
  r.q[1] = *(const uint4*)p1;
  return r.v;
}
// B-operand (32x16 bf16): per lane two 16B chunks: K = g*8..g*8+7 and +16..+23
__device__ __forceinline__ v16bf ld_b_op(const __bf16* p) {
  BFPack r;
  r.q[0] = *(const uint4*)p;
  r.q[1] = *(const uint4*)(p + 16);
  return r.v;
}

__device__ __forceinline__ v8f wmma_bf16(v16bf a, v16bf b, v8f c) {
  // (neg_a, A, neg_b, B, c_mod, C, reuse_a, reuse_b)
  return __builtin_amdgcn_wmma_f32_16x16x32_bf16(false, a, false, b,
                                                 (short)0, c, false, false);
}

// reductions across the 16-lane half-groups (wave32; masks 1/2/4/8 stay in-group)
__device__ __forceinline__ float grp_max(float v) {
  v = fmaxf(v, __shfl_xor(v, 1, 32));
  v = fmaxf(v, __shfl_xor(v, 2, 32));
  v = fmaxf(v, __shfl_xor(v, 4, 32));
  v = fmaxf(v, __shfl_xor(v, 8, 32));
  return v;
}
__device__ __forceinline__ float grp_sum(float v) {
  v += __shfl_xor(v, 1, 32);
  v += __shfl_xor(v, 2, 32);
  v += __shfl_xor(v, 4, 32);
  v += __shfl_xor(v, 8, 32);
  return v;
}

// ------------------------------ elementwise converts ------------------------
__global__ void __launch_bounds__(256)
cvt_bf16_k(const float* __restrict__ in, __bf16* __restrict__ out, int n) {
  int i = blockIdx.x * 256 + threadIdx.x;
  if (i < n) out[i] = (__bf16)in[i];
}

__global__ void __launch_bounds__(256)
tcvt_k(const float* __restrict__ W, __bf16* __restrict__ WT) {
  int idx = blockIdx.x * 256 + threadIdx.x;   // 512*512 threads
  int n = idx & (D_MODEL - 1);
  int k = idx >> 9;
  WT[(size_t)n * D_MODEL + k] = (__bf16)W[(size_t)k * D_MODEL + n];
}

// ------------------------------ WMMA GEMM -----------------------------------
// C[M,512] = A[M,512] @ BT[512,512]^T + bias ; one wave = 16x64 output tile.
// mode 1: bf16 out at [B,H,S,Hd]   (q, k)
// mode 2: bf16 out at [B,H,Hd,S]   (v transposed)
// mode 3: f32  out at [M,512]      (final projection)
__global__ void __launch_bounds__(128)
gemm_bf16_k(const __bf16* __restrict__ A, const __bf16* __restrict__ BT,
            const float* __restrict__ bias, void* __restrict__ out, int mode) {
  const int lane = threadIdx.x & 31;
  const int li = lane & 15, g = lane >> 4;
  const int gw = blockIdx.x * 4 + (threadIdx.x >> 5);
  const int m0 = (gw >> 3) * 16;
  const int n0 = (gw & 7) * 64;

  v8f acc[4] = {{}, {}, {}, {}};
  const __bf16* arow  = A + (size_t)(m0 + li) * D_MODEL;
  const __bf16* brow  = BT + (size_t)(n0 + li) * D_MODEL;   // + t*16*D_MODEL per tile

#pragma unroll 2
  for (int kk = 0; kk < D_MODEL; kk += 32) {
    // all loads into distinct registers first, so loadcnt waits stagger
    const __bf16* pa = arow + kk + g * 8;
    v16bf av = ld_a_op(pa, pa + 16);
    v16bf bv[4];
#pragma unroll
    for (int t = 0; t < 4; ++t)
      bv[t] = ld_b_op(brow + (size_t)t * 16 * D_MODEL + kk + g * 8);
#pragma unroll
    for (int t = 0; t < 4; ++t)
      acc[t] = wmma_bf16(av, bv[t], acc[t]);
  }

#pragma unroll
  for (int t = 0; t < 4; ++t) {
#pragma unroll
    for (int r = 0; r < 8; ++r) {
      const int gm = m0 + r + 8 * g;            // global row (b*S + s)
      const int n  = n0 + t * 16 + li;          // global col (h*64 + d)
      float val = acc[t][r] + bias[n];
      if (mode == 1) {
        int b = gm >> 10, s = gm & (SEQ - 1);
        int h = n >> 6,   d = n & (HD - 1);
        ((__bf16*)out)[(((size_t)(b * NH + h) * SEQ) + s) * HD + d] = (__bf16)val;
      } else if (mode == 2) {
        int b = gm >> 10, s = gm & (SEQ - 1);
        int h = n >> 6,   d = n & (HD - 1);
        ((__bf16*)out)[(((size_t)(b * NH + h) * HD) + d) * SEQ + s] = (__bf16)val;
      } else {
        ((float*)out)[(size_t)gm * D_MODEL + n] = val;
      }
    }
  }
}

// ------------------------------ temporal-bias MLP ---------------------------
// bias8[b*8+h][i][j] = relu(td*W1+b1) @ W2[:,h] + b2[h]   (bf16 out)
__global__ void __launch_bounds__(256)
bias_mlp_k(const float* __restrict__ td, const float* __restrict__ Wt1,
           const float* __restrict__ bt1, const float* __restrict__ Wt2,
           const float* __restrict__ bt2, __bf16* __restrict__ bias8) {
  __shared__ float w1[32], b1[32], w2[256], b2[8];
  const int tid = threadIdx.x;
  if (tid < 32) { w1[tid] = Wt1[tid]; b1[tid] = bt1[tid]; }
  w2[tid] = Wt2[tid];
  if (tid < 8) b2[tid] = bt2[tid];
  __syncthreads();

  const size_t idx = (size_t)blockIdx.x * 256 + tid;   // (b*S + i)*S + j
  const float x = td[idx];
  float o[8];
#pragma unroll
  for (int h = 0; h < 8; ++h) o[h] = b2[h];
#pragma unroll
  for (int c = 0; c < 32; ++c) {
    float hv = fmaxf(fmaf(x, w1[c], b1[c]), 0.0f);
#pragma unroll
    for (int h = 0; h < 8; ++h) o[h] = fmaf(hv, w2[c * 8 + h], o[h]);
  }
  const int j = (int)(idx & (SEQ - 1));
  const int i = (int)((idx >> 10) & (SEQ - 1));
  const int b = (int)(idx >> 20);
#pragma unroll
  for (int h = 0; h < 8; ++h)
    bias8[(((size_t)(b * NH + h) * SEQ + i) * SEQ) + j] = (__bf16)o[h];
}

// ------------------------------ flash attention -----------------------------
// One wave: 16 query rows of one (b,h); streams j in steps of 32.
// Independent loads (V tiles, bias) are hoisted to the top of each iteration
// so they are in flight underneath the score WMMAs and softmax VALU work.
__global__ void __launch_bounds__(128)
attn_k(const __bf16* __restrict__ qh, const __bf16* __restrict__ kh,
       const __bf16* __restrict__ vT, const __bf16* __restrict__ bias8,
       __bf16* __restrict__ ctx) {
  __shared__ __bf16 plds[4][16][40];               // per-wave P relayout buffer
  const int lane = threadIdx.x & 31;
  const int wid  = threadIdx.x >> 5;
  const int li = lane & 15, g = lane >> 4;
  const int gw = blockIdx.x * 4 + wid;             // 0..2047
  const int bh = gw >> 6;                          // b*8 + h
  const int i0 = (gw & 63) * 16;

  // Q tile as A-operand for both K-steps (d 0..31, 32..63)
  const __bf16* pq = qh + ((size_t)bh * SEQ + i0 + li) * HD + g * 8;
  const v16bf qa0 = ld_a_op(pq,      pq + 16);
  const v16bf qa1 = ld_a_op(pq + 32, pq + 48);

  v8f O[4] = {{}, {}, {}, {}};
  float mrow[8], lrow[8];
#pragma unroll
  for (int r = 0; r < 8; ++r) { mrow[r] = -1.0e30f; lrow[r] = 0.0f; }

  const __bf16* kbase = kh + (size_t)bh * SEQ * HD;
  const __bf16* vbase = vT + (size_t)bh * HD * SEQ;
  const __bf16* bbase = bias8 + ((size_t)bh * SEQ + i0) * SEQ;

  for (int jb = 0; jb < SEQ; jb += 32) {
    // ---- independent loads first: V tiles, bias, K tiles (distinct regs)
    v16bf vb[4];
#pragma unroll
    for (int t = 0; t < 4; ++t)
      vb[t] = ld_b_op(vbase + (size_t)(t * 16 + li) * SEQ + jb + g * 8);

    float bv0[8], bv1[8];
#pragma unroll
    for (int r = 0; r < 8; ++r) {
      const int mloc = r + 8 * g;
      bv0[r] = (float)bbase[(size_t)mloc * SEQ + jb + li];
      bv1[r] = (float)bbase[(size_t)mloc * SEQ + jb + 16 + li];
    }

    const __bf16* pk0 = kbase + (size_t)(jb + li) * HD + g * 8;
    const __bf16* pk1 = pk0 + 16 * HD;
    v16bf k0lo = ld_b_op(pk0);
    v16bf k0hi = ld_b_op(pk0 + 32);
    v16bf k1lo = ld_b_op(pk1);
    v16bf k1hi = ld_b_op(pk1 + 32);

    // prefetch next iteration's K rows into cache (global_prefetch_b8)
    if (jb + 32 < SEQ)
      __builtin_prefetch(kbase + (size_t)(jb + 32 + li) * HD, 0, 3);

    // ---- scores: two 16x16 j-tiles, K=64 as two wmma steps each
    v8f s0 = {}, s1 = {};
    s0 = wmma_bf16(qa0, k0lo, s0);
    s0 = wmma_bf16(qa1, k0hi, s0);
    s1 = wmma_bf16(qa0, k1lo, s1);
    s1 = wmma_bf16(qa1, k1hi, s1);

    // ---- scale + temporal bias (C/D layout: row = r+8g, col = li / li+16)
    float p0[8], p1[8];
#pragma unroll
    for (int r = 0; r < 8; ++r) {
      p0[r] = s0[r] * 0.125f + bv0[r];
      p1[r] = s1[r] * 0.125f + bv1[r];
    }

    // ---- online softmax + stage P into LDS (bf16, A-operand friendly)
#pragma unroll
    for (int r = 0; r < 8; ++r) {
      float nm = fmaxf(mrow[r], grp_max(fmaxf(p0[r], p1[r])));
      float alpha = __expf(mrow[r] - nm);
      mrow[r] = nm;
      p0[r] = __expf(p0[r] - nm);
      p1[r] = __expf(p1[r] - nm);
      lrow[r] = lrow[r] * alpha + grp_sum(p0[r] + p1[r]);
      O[0][r] *= alpha; O[1][r] *= alpha; O[2][r] *= alpha; O[3][r] *= alpha;
      plds[wid][r + 8 * g][li]      = (__bf16)p0[r];
      plds[wid][r + 8 * g][li + 16] = (__bf16)p1[r];
    }
    asm volatile("s_wait_dscnt 0x0" ::: "memory");

    // ---- P (16x32) as A-operand; V tiles already resident
    const __bf16* pp = &plds[wid][li][g * 8];
    v16bf pa = ld_a_op(pp, pp + 16);
#pragma unroll
    for (int t = 0; t < 4; ++t)
      O[t] = wmma_bf16(pa, vb[t], O[t]);
  }

  // ---- normalize and store context at [B,S,D] (D index = h*64 + d)
  const int b = bh >> 3, h = bh & 7;
#pragma unroll
  for (int r = 0; r < 8; ++r) {
    const float inv = 1.0f / lrow[r];
    const int srow = i0 + r + 8 * g;
#pragma unroll
    for (int t = 0; t < 4; ++t)
      ctx[((size_t)(b * SEQ + srow)) * D_MODEL + h * HD + t * 16 + li] =
          (__bf16)(O[t][r] * inv);
  }
}

// ------------------------------ launcher ------------------------------------
extern "C" void kernel_launch(void* const* d_in, const int* in_sizes, int n_in,
                              void* d_out, int out_size, void* d_ws, size_t ws_size,
                              hipStream_t stream) {
  (void)in_sizes; (void)n_in; (void)out_size; (void)ws_size;
  const float* query = (const float*)d_in[0];
  const float* key_  = (const float*)d_in[1];
  const float* value = (const float*)d_in[2];
  const float* tdel  = (const float*)d_in[3];
  const float* Wq = (const float*)d_in[4];
  const float* bq = (const float*)d_in[5];
  const float* Wk = (const float*)d_in[6];
  const float* bk = (const float*)d_in[7];
  const float* Wv = (const float*)d_in[8];
  const float* bv = (const float*)d_in[9];
  const float* Wo = (const float*)d_in[10];
  const float* bo = (const float*)d_in[11];
  const float* Wt1 = (const float*)d_in[12];
  const float* bt1 = (const float*)d_in[13];
  const float* Wt2 = (const float*)d_in[14];
  const float* bt2 = (const float*)d_in[15];

  const size_t NX = (size_t)NBATCH * SEQ * D_MODEL;   // 2,097,152
  const size_t NW = (size_t)D_MODEL * D_MODEL;        //   262,144

  __bf16* p = (__bf16*)d_ws;
  __bf16* xq  = p; p += NX;
  __bf16* xk  = p; p += NX;
  __bf16* xv  = p; p += NX;
  __bf16* wqt = p; p += NW;
  __bf16* wkt = p; p += NW;
  __bf16* wvt = p; p += NW;
  __bf16* wot = p; p += NW;
  __bf16* qh  = p; p += NX;
  __bf16* kh  = p; p += NX;
  __bf16* vt  = p; p += NX;
  __bf16* bias8 = p; p += (size_t)NBATCH * NH * SEQ * SEQ;  // 64 MB
  __bf16* ctx = p; p += NX;

  cvt_bf16_k<<<8192, 256, 0, stream>>>(query, xq, (int)NX);
  cvt_bf16_k<<<8192, 256, 0, stream>>>(key_,  xk, (int)NX);
  cvt_bf16_k<<<8192, 256, 0, stream>>>(value, xv, (int)NX);
  tcvt_k<<<1024, 256, 0, stream>>>(Wq, wqt);
  tcvt_k<<<1024, 256, 0, stream>>>(Wk, wkt);
  tcvt_k<<<1024, 256, 0, stream>>>(Wv, wvt);
  tcvt_k<<<1024, 256, 0, stream>>>(Wo, wot);

  gemm_bf16_k<<<512, 128, 0, stream>>>(xq, wqt, bq, qh, 1);
  gemm_bf16_k<<<512, 128, 0, stream>>>(xk, wkt, bk, kh, 1);
  gemm_bf16_k<<<512, 128, 0, stream>>>(xv, wvt, bv, vt, 2);

  bias_mlp_k<<<16384, 256, 0, stream>>>(tdel, Wt1, bt1, Wt2, bt2, bias8);

  attn_k<<<512, 128, 0, stream>>>(qh, kh, vt, bias8, ctx);

  gemm_bf16_k<<<512, 128, 0, stream>>>(ctx, wot, bo, d_out, 3);
}
